// Attention_43954695307828
// MI455X (gfx1250) — matrix operand order repaired and checked by reading, not compiled, not run on hardware
//
#include <hip/hip_runtime.h>

// Problem constants (from reference): B=2, S=2048, D=4096, NH=32, NKV=8, HD=128
#define Bsz 2
#define Ssz 2048
#define Dsz 4096
#define NHsz 32
#define NKVsz 8
#define HDsz 128

typedef __attribute__((ext_vector_type(16))) __bf16 bf16x16;
typedef __attribute__((ext_vector_type(8)))  float  floatx8;
typedef __attribute__((ext_vector_type(4)))  unsigned int uint4v;
typedef __attribute__((ext_vector_type(2)))  unsigned int uint2v;
typedef __attribute__((ext_vector_type(4)))  unsigned int u32x4;
typedef __attribute__((ext_vector_type(8)))  int          i32x8;
typedef __attribute__((ext_vector_type(4)))  int          i32x4;

union FragU {
    uint4v  u4[2];
    uint2v  u2[4];
    bf16x16 f;
};

__device__ __forceinline__ unsigned short f2bf(float x) {
    unsigned int u = __float_as_uint(x);
    u += 0x7fffu + ((u >> 16) & 1u);   // round-to-nearest-even
    return (unsigned short)(u >> 16);
}

__device__ __forceinline__ floatx8 wmma_bf16(bf16x16 a, bf16x16 b, floatx8 c) {
    // (neg_a, A, neg_b, B, c_mod, C, reuse_a, reuse_b)
    return __builtin_amdgcn_wmma_f32_16x16x32_bf16(false, a, false, b, (short)0, c, false, false);
}

// A fragment: 16x32 bf16 tile from row-major src (already offset to tile origin),
// leading dim ld (elements). ISA layout: lane<16 row=lane K-base=0, lane>=16 K-base=8;
// VGPR0-3: K=kb..kb+7, VGPR4-7: K=16+kb..16+kb+7.
__device__ __forceinline__ bf16x16 load_frag_a(const unsigned short* src, int ld, int lane) {
    FragU f;
    const int r  = lane & 15;
    const int kb = (lane >> 4) * 8;
    const unsigned short* p = src + (size_t)r * ld + kb;
    f.u4[0] = *(const uint4v*)p;          // K = kb .. kb+7
    f.u4[1] = *(const uint4v*)(p + 16);   // K = 16+kb .. 16+kb+7
    return f.f;
}

// B fragment: 32x16 tile given rows of B^T (row-major [N,K], already offset).
// ISA layout: lane<16 col=lane K-base=0, lane>=16 col=lane-16 K-base=8; VGPR0-3: K=kb..kb+7, VGPR4-7: K=16+kb..16+kb+7.
__device__ __forceinline__ bf16x16 load_frag_b(const unsigned short* srcT, int ld, int lane) {
    FragU f;
    const int n  = lane & 15;
    const int kb = (lane >> 4) * 8;
    const unsigned short* p = srcT + (size_t)n * ld + kb;
    f.u4[0] = *(const uint4v*)p;          // K = kb .. kb+7
    f.u4[1] = *(const uint4v*)(p + 16);   // K = 16+kb .. 16+kb+7
    return f.f;
}

__device__ __forceinline__ void store_out(float* p, float v)          { *p = v; }
__device__ __forceinline__ void store_out(unsigned short* p, float v) { *p = f2bf(v); }

// ---------------- Tensor Data Mover: 2D bf16 tile -> LDS ----------------
// D# per CDNA5 ISA ch.8: group0 = {flags, lds_addr, global_addr}, group1 = dims/strides.
// Loads a (tile_d1 rows x tile_d0 elems) tile of a [tensor_d1, tensor_d0] bf16 tensor.
// Pad config (in g1flags): after every 16 dwords (32 elems) insert 4 dwords (8 elems),
// producing a 40-element LDS row stride from a 32-element tile row.
__device__ __forceinline__ void tdm_load_2d(unsigned lds_byte_addr,
                                            unsigned long long gaddr,
                                            unsigned tensor_d0, unsigned tensor_d1,
                                            unsigned tile_d0, unsigned tile_d1,
                                            unsigned stride0_elems,
                                            int g1flags) {
    u32x4 g0;
    g0[0] = 1u;                                   // count=1, user descriptor
    g0[1] = lds_byte_addr;                        // D#.lds_addr (bytes)
    g0[2] = (unsigned)gaddr;                      // global_addr[31:0]
    g0[3] = (unsigned)((gaddr >> 32) & 0x1FFFFFFull) | (2u << 30);  // addr[56:32] | type=2
    i32x8 g1;
    g1[0] = g1flags;                                              // data_size | pad cfg
    g1[1] = (int)((tensor_d0 & 0xFFFFu) << 16);                   // tensor_dim0[15:0]
    g1[2] = (int)(((tensor_d0 >> 16) & 0xFFFFu) | ((tensor_d1 & 0xFFFFu) << 16));
    g1[3] = (int)(((tensor_d1 >> 16) & 0xFFFFu) | (tile_d0 << 16));
    g1[4] = (int)(tile_d1 & 0xFFFFu);                             // tile_dim1, tile_dim2=0
    g1[5] = (int)stride0_elems;                                   // tensor_dim0_stride lo
    g1[6] = 0;
    g1[7] = 0;
    i32x4 zz = {0, 0, 0, 0};                       // groups 2/3 unused (2D tensor)
    i32x8 z8 = {0, 0, 0, 0, 0, 0, 0, 0};           // extra group (clang-23 6-arg form)
    __builtin_amdgcn_tensor_load_to_lds(g0, g1, zz, zz, z8, 0);
}

// data_size=2B (code 1), pad_enable, pad_interval=16 dwords (code 3), pad_amount=4 dwords (code 3)
#define TDM_G1_FLAGS ((1 << 16) | (1 << 20) | (3 << 22) | (3 << 25))
#define LDS_STRIDE 40   // 32 tile elems + 8 pad elems; 80B rows keep b128 alignment

// ---------------- conversion kernels ----------------

__global__ __launch_bounds__(256) void cvt_bf16_kernel(const float* __restrict__ in,
                                                       unsigned short* __restrict__ out, int n) {
    int i = blockIdx.x * 256 + threadIdx.x;
    if (i < n) out[i] = f2bf(in[i]);
}

// w [K,N] fp32 row-major -> wT [N,K] bf16 row-major
__global__ __launch_bounds__(256) void cvt_transpose_kernel(const float* __restrict__ w,
                                                            unsigned short* __restrict__ wT,
                                                            int K, int N) {
    int i = blockIdx.x * 256 + threadIdx.x;
    if (i < K * N) {
        int k = i / N, n = i - k * N;
        wT[(size_t)n * K + k] = f2bf(w[i]);
    }
}

// v [B*S, NKV*HD] bf16 -> vT [B][NKV][HD][S] bf16
__global__ __launch_bounds__(256) void transpose_v_kernel(const unsigned short* __restrict__ v,
                                                          unsigned short* __restrict__ vT, int n) {
    int i = blockIdx.x * 256 + threadIdx.x;
    if (i >= n) return;
    int row = i >> 10, col = i & 1023;          // 1024 = NKV*HD
    int b = row >> 11, s = row & 2047;          // S = 2048
    int g = col >> 7,  d = col & 127;           // HD = 128
    vT[(((size_t)(b * NKVsz + g) * HDsz + d) << 11) + s] = v[i];
}

// ---------------- TDM-fed WMMA GEMM: C[M,N] = A[M,K] * Bt[N,K]^T ----------------
// block = 256 threads (8 waves), block tile 128(M) x 128(N), K-chunk 32.
// Wave 0 drives the Tensor Data Mover into double-buffered padded LDS tiles;
// all 8 waves consume via ds_load_b128 fragments (wave tile 32 x 64).

template <typename OutT>
__global__ __launch_bounds__(256) void gemm_bt_kernel(const unsigned short* __restrict__ A,
                                                      const unsigned short* __restrict__ Bt,
                                                      OutT* __restrict__ C,
                                                      int M, int N, int K) {
    __shared__ unsigned short ldsA[2][128 * LDS_STRIDE];
    __shared__ unsigned short ldsB[2][128 * LDS_STRIDE];

    const int lane = threadIdx.x & 31;
    const int wid  = threadIdx.x >> 5;
    const int wm = wid & 3;            // 4 M-strips of 32
    const int wn = wid >> 2;           // 2 N-strips of 64
    const int M0 = blockIdx.y * 128;
    const int N0 = blockIdx.x * 128;

    const floatx8 z = {0.f,0.f,0.f,0.f,0.f,0.f,0.f,0.f};
    floatx8 acc[2][4];
#pragma unroll
    for (int i = 0; i < 2; ++i)
#pragma unroll
        for (int j = 0; j < 4; ++j) acc[i][j] = z;

    // prologue: DMA first K-chunk into buffer 0 (TDM ignores EXEC; one wave issues)
    if (wid == 0) {
        tdm_load_2d((unsigned)(unsigned long long)&ldsA[0][0],
                    (unsigned long long)(const void*)(A + (size_t)M0 * K),
                    (unsigned)K, (unsigned)M, 32u, 128u, (unsigned)K, TDM_G1_FLAGS);
        tdm_load_2d((unsigned)(unsigned long long)&ldsB[0][0],
                    (unsigned long long)(const void*)(Bt + (size_t)N0 * K),
                    (unsigned)K, (unsigned)N, 32u, 128u, (unsigned)K, TDM_G1_FLAGS);
    }

    int buf = 0;
    for (int k0 = 0; k0 < K; k0 += 32) {
        if (wid == 0) {
            if (k0 + 32 < K) {   // prefetch next chunk into the other buffer
                tdm_load_2d((unsigned)(unsigned long long)&ldsA[buf ^ 1][0],
                            (unsigned long long)(const void*)(A + (size_t)M0 * K + k0 + 32),
                            (unsigned)K, (unsigned)M, 32u, 128u, (unsigned)K, TDM_G1_FLAGS);
                tdm_load_2d((unsigned)(unsigned long long)&ldsB[buf ^ 1][0],
                            (unsigned long long)(const void*)(Bt + (size_t)N0 * K + k0 + 32),
                            (unsigned)K, (unsigned)N, 32u, 128u, (unsigned)K, TDM_G1_FLAGS);
                __builtin_amdgcn_s_wait_tensorcnt(2);   // current chunk has landed
            } else {
                __builtin_amdgcn_s_wait_tensorcnt(0);
            }
        }
        __syncthreads();   // data for this chunk visible to all waves

        bf16x16 a0 = load_frag_a(&ldsA[buf][(wm * 32) * LDS_STRIDE],      LDS_STRIDE, lane);
        bf16x16 a1 = load_frag_a(&ldsA[buf][(wm * 32 + 16) * LDS_STRIDE], LDS_STRIDE, lane);
#pragma unroll
        for (int j = 0; j < 4; ++j) {
            bf16x16 bf = load_frag_b(&ldsB[buf][(wn * 64 + 16 * j) * LDS_STRIDE], LDS_STRIDE, lane);
            acc[0][j] = wmma_bf16(a0, bf, acc[0][j]);
            acc[1][j] = wmma_bf16(a1, bf, acc[1][j]);
        }

        __syncthreads();   // all waves done reading before buffer is re-filled
        buf ^= 1;
    }

    const int colb = lane & 15;
    const int hi   = lane >> 4;
    const int m0 = M0 + wm * 32;
    const int n0 = N0 + wn * 64;
#pragma unroll
    for (int i = 0; i < 2; ++i)
#pragma unroll
        for (int j = 0; j < 4; ++j)
#pragma unroll
            for (int r = 0; r < 8; ++r) {
                int row = m0 + 16 * i + r + 8 * hi;
                int col = n0 + 16 * j + colb;
                store_out(C + (size_t)row * N + col, acc[i][j][r]);
            }
}

// ---------------- Flash attention ----------------
// One wave = 16 query rows of one head. 256 threads = 8 independent waves per block.

__global__ __launch_bounds__(256) void attn_kernel(const unsigned short* __restrict__ Q,  // [B*S, NH*HD]
                                                   const unsigned short* __restrict__ Kc, // [B*S, NKV*HD]
                                                   const unsigned short* __restrict__ Vt, // [B][NKV][HD][S]
                                                   unsigned short* __restrict__ O) {      // [B*S, NH*HD]
    __shared__ unsigned short plds[8][16 * 40];   // per-wave P transpose staging (stride 40)

    const int lane = threadIdx.x & 31;
    const int wid  = threadIdx.x >> 5;
    const int gw   = blockIdx.x * 8 + wid;
    const int qt = gw & 127;            // S/16 = 128 query tiles
    const int h  = (gw >> 7) & 31;      // head
    const int b  = gw >> 12;            // batch
    const int g  = h >> 2;              // kv head (N_REP = 4)
    const float scale = 0.08838834764831845f;  // 1/sqrt(128)

    const int qrow0 = b * Ssz + qt * 16;
    const int hi    = lane >> 4;
    const int colb  = lane & 15;

    bf16x16 qa[4];
#pragma unroll
    for (int c = 0; c < 4; ++c)
        qa[c] = load_frag_a(Q + (size_t)qrow0 * Dsz + h * HDsz + c * 32, Dsz, lane);

    const floatx8 z = {0.f,0.f,0.f,0.f,0.f,0.f,0.f,0.f};
    floatx8 acc[8];
#pragma unroll
    for (int t = 0; t < 8; ++t) acc[t] = z;
    float mrow[8], lrow[8];
#pragma unroll
    for (int r = 0; r < 8; ++r) { mrow[r] = -3.0e38f; lrow[r] = 0.f; }

    const unsigned short* kbase = Kc + (size_t)b * Ssz * 1024 + g * HDsz;
    const unsigned short* vbase = Vt + (size_t)(b * NKVsz + g) * HDsz * Ssz;
    unsigned short* myp = &plds[wid][0];

    for (int s0 = 0; s0 < Ssz; s0 += 32) {
        floatx8 sc0 = z, sc1 = z;
#pragma unroll
        for (int c = 0; c < 4; ++c) {
            bf16x16 kf0 = load_frag_b(kbase + (size_t)s0 * 1024 + c * 32, 1024, lane);
            bf16x16 kf1 = load_frag_b(kbase + (size_t)(s0 + 16) * 1024 + c * 32, 1024, lane);
            sc0 = wmma_bf16(qa[c], kf0, sc0);
            sc1 = wmma_bf16(qa[c], kf1, sc1);
        }
        sc0 *= scale;
        sc1 *= scale;

#pragma unroll
        for (int r = 0; r < 8; ++r) {
            // row max over the 32 kv columns (16 lanes hold 16 cols of each tile)
            float v = fmaxf(sc0[r], sc1[r]);
#pragma unroll
            for (int msk = 1; msk < 16; msk <<= 1) v = fmaxf(v, __shfl_xor(v, msk, 32));
            float mnew  = fmaxf(mrow[r], v);
            float alpha = __expf(mrow[r] - mnew);
            mrow[r] = mnew;

            float p0 = __expf(sc0[r] - mnew);
            float p1 = __expf(sc1[r] - mnew);
            float rs = p0 + p1;
#pragma unroll
            for (int msk = 1; msk < 16; msk <<= 1) rs += __shfl_xor(rs, msk, 32);
            lrow[r] = lrow[r] * alpha + rs;

            int prow = r + 8 * hi;
            myp[prow * 40 + colb]      = f2bf(p0);
            myp[prow * 40 + 16 + colb] = f2bf(p1);
#pragma unroll
            for (int t = 0; t < 8; ++t) acc[t][r] *= alpha;
        }

        // P: C-layout -> A-fragment via LDS (8B-aligned ds loads, stride 40 elems)
        FragU pf;
        {
            const int r2 = lane & 15;
            const int kb = hi * 8;
            const unsigned short* p = myp + r2 * 40 + kb;
            pf.u2[0] = *(const uint2v*)p;
            pf.u2[1] = *(const uint2v*)(p + 4);
            pf.u2[2] = *(const uint2v*)(p + 16);
            pf.u2[3] = *(const uint2v*)(p + 20);
        }

#pragma unroll
        for (int t = 0; t < 8; ++t) {
            bf16x16 vf = load_frag_b(vbase + (size_t)(t * 16) * Ssz + s0, Ssz, lane);
            acc[t] = wmma_bf16(pf.f, vf, acc[t]);
        }
    }

#pragma unroll
    for (int r = 0; r < 8; ++r) {
        float inv = 1.0f / lrow[r];
        int row = qrow0 + r + 8 * hi;
#pragma unroll
        for (int t = 0; t < 8; ++t) {
            O[(size_t)row * Dsz + h * HDsz + t * 16 + colb] = f2bf(acc[t][r] * inv);
        }
    }
}

// ---------------- launcher ----------------

extern "C" void kernel_launch(void* const* d_in, const int* in_sizes, int n_in,
                              void* d_out, int out_size, void* d_ws, size_t ws_size,
                              hipStream_t stream) {
    (void)in_sizes; (void)n_in; (void)out_size; (void)ws_size;
    const float* x  = (const float*)d_in[0];
    const float* wq = (const float*)d_in[1];
    const float* wk = (const float*)d_in[2];
    const float* wv = (const float*)d_in[3];
    const float* wo = (const float*)d_in[4];

    char* ws = (char*)d_ws;
    const size_t MB = 1024 * 1024;
    unsigned short* xbf = (unsigned short*)(ws +   0 * MB);  // 32MB (aliased by attn later)
    unsigned short* wqT = (unsigned short*)(ws +  32 * MB);  // 32MB (aliased by woT later)
    unsigned short* wkT = (unsigned short*)(ws +  64 * MB);  //  8MB
    unsigned short* wvT = (unsigned short*)(ws +  72 * MB);  //  8MB
    unsigned short* qbf = (unsigned short*)(ws +  80 * MB);  // 32MB
    unsigned short* kbf = (unsigned short*)(ws + 112 * MB);  //  8MB
    unsigned short* vbf = (unsigned short*)(ws + 120 * MB);  //  8MB
    unsigned short* vT  = (unsigned short*)(ws + 128 * MB);  //  8MB
    unsigned short* attn = xbf;   // x no longer needed after QKV GEMMs
    unsigned short* woT  = wqT;   // wq no longer needed after Q GEMM

    // bf16 conversions (weights transposed to [N,K] for contiguous B fragments)
    cvt_bf16_kernel<<<65536, 256, 0, stream>>>(x, xbf, Bsz * Ssz * Dsz);
    cvt_transpose_kernel<<<65536, 256, 0, stream>>>(wq, wqT, Dsz, Dsz);
    cvt_transpose_kernel<<<16384, 256, 0, stream>>>(wk, wkT, Dsz, NKVsz * HDsz);
    cvt_transpose_kernel<<<16384, 256, 0, stream>>>(wv, wvT, Dsz, NKVsz * HDsz);

    // QKV projections (M = B*S = 4096, K = 4096), TDM-fed 128x128 block tiles
    gemm_bt_kernel<unsigned short><<<dim3(4096 / 128, 4096 / 128), 256, 0, stream>>>(
        xbf, wqT, qbf, 4096, 4096, 4096);
    gemm_bt_kernel<unsigned short><<<dim3(1024 / 128, 4096 / 128), 256, 0, stream>>>(
        xbf, wkT, kbf, 4096, 1024, 4096);
    gemm_bt_kernel<unsigned short><<<dim3(1024 / 128, 4096 / 128), 256, 0, stream>>>(
        xbf, wvT, vbf, 4096, 1024, 4096);

    // wo conversion (reuses wq scratch), V transpose for PV fragments
    cvt_transpose_kernel<<<65536, 256, 0, stream>>>(wo, woT, Dsz, Dsz);
    transpose_v_kernel<<<16384, 256, 0, stream>>>(vbf, vT, Bsz * Ssz * NKVsz * HDsz);

    // flash attention: 8192 waves (B*NH*S/16), 8 waves per block
    attn_kernel<<<1024, 256, 0, stream>>>(qbf, kbf, vT, attn);

    // output projection -> fp32
    gemm_bt_kernel<float><<<dim3(4096 / 128, 4096 / 128), 256, 0, stream>>>(
        attn, woT, (float*)d_out, 4096, 4096, 4096);
}
